// TemporalAttention4_55138790146545
// MI455X (gfx1250) — matrix operand order repaired and checked by reading, not compiled, not run on hardware
//
#include <hip/hip_runtime.h>

typedef __attribute__((ext_vector_type(16))) _Float16 v16h;
typedef __attribute__((ext_vector_type(8)))  _Float16 v8h;
typedef __attribute__((ext_vector_type(8)))  float    v8f;

#define T_SEQ 2048
#define D_DIM 512
#define W_TOP 12
#define NROW  8192        // B * (T/4) = 16 * 512
#define LDST  520         // padded LDS row stride in halves (breaks bank pattern)
#define NCT   96          // column tiles per weight matrix (1536 / 16)
#define NKC   16          // k-chunks (512 / 32)

// A-fragment (16x32 f16, M x K) from a hoisted per-lane LDS base.
// ISA layout: lane l -> M = l&15, group g = l>>4; per-lane K values:
// p<8 -> k = g*8+p, p>=8 -> k = 16+g*8+(p-8) => two 16B runs at +k0, +k0+16.
__device__ inline v16h ldA(const _Float16* base, int k0) {
  union { v16h v; v8h h[2]; } u;
  u.h[0] = *(const v8h*)(base + k0);
  u.h[1] = *(const v8h*)(base + k0 + 16);
  return u.v;
}

// ---------------------------------------------------------------------------
// Kernel 1: f32 -> f16 weight convert + swizzle into WMMA B-fragment order.
// Fragment (ct, kc) stored contiguously (1KB) in [lane][p] order:
//   element = w[ct*16 + (l&15)][kc*32 + (l>>4)*8 + (p<8 ? p : p+8)]
// ---------------------------------------------------------------------------
__global__ __launch_bounds__(256) void pack_w_kernel(
    const float* __restrict__ wi, const float* __restrict__ wh,
    _Float16* __restrict__ wiP, _Float16* __restrict__ whP) {
  int fi = blockIdx.x * 256 + threadIdx.x;      // 0 .. 96*16*32-1
  if (fi >= NCT * NKC * 32) return;
  int l  = fi & 31;
  int kc = (fi >> 5) & (NKC - 1);
  int ct = fi >> 9;
  int row = ct * 16 + (l & 15);                 // output column n
  int k0  = kc * 32 + (l >> 4) * 8;             // first k of this lane
  const float* si = wi + (size_t)row * D_DIM + k0;
  const float* sh = wh + (size_t)row * D_DIM + k0;
  size_t dst = ((size_t)(((ct << 4) + kc) << 5) + l) * 16;
#pragma unroll
  for (int p = 0; p < 16; ++p) {
    wiP[dst + p] = (_Float16)si[p < 8 ? p : p + 8];
    whP[dst + p] = (_Float16)sh[p < 8 ? p : p + 8];
  }
}

// ---------------------------------------------------------------------------
// Kernel 2: banded top-12 neighbor selection, fp32 exact (selection must
// match the fp32 reference ranking; band has <=23, >=12 candidates).
// One wave32 per query; lane = candidate. Rank via shuffles, compact the
// selected (already index-ascending) set with ballot+popc.
// ---------------------------------------------------------------------------
__global__ __launch_bounds__(256) void topk_kernel(
    const float* __restrict__ x, int* __restrict__ ids) {
  int wave = threadIdx.x >> 5;
  int lane = threadIdx.x & 31;
  int row  = blockIdx.x * 8 + wave;       // 0..8191
  int b    = row >> 9;
  int qi   = row & 511;
  int tq   = (qi * 2047) / 511;           // matches np.linspace(...).astype(i32)
  int lo   = tq - (W_TOP - 1); if (lo < 0) lo = 0;
  int hi   = tq + (W_TOP - 1); if (hi > T_SEQ - 1) hi = T_SEQ - 1;
  int nc   = hi - lo + 1;                 // 12..23
  int j    = lo + lane;
  bool valid = lane < nc;
  int jc   = valid ? j : tq;              // safe address for dead lanes

  const float* q = x + ((size_t)b * T_SEQ + tq) * D_DIM;
  const float* c = x + ((size_t)b * T_SEQ + jc) * D_DIM;
  float dot = 0.f;
  for (int k = 0; k < D_DIM; k += 4) {
    float4 qv = *(const float4*)(q + k);
    float4 cv = *(const float4*)(c + k);
    dot += qv.x * cv.x + qv.y * cv.y + qv.z * cv.z + qv.w * cv.w;
  }
  float s = valid ? dot : -3.0e38f;

  // rank = #candidates strictly better (tie -> lower index wins, = jax top_k)
  int rank = 0;
#pragma unroll
  for (int m = 0; m < 32; ++m) {
    float sm = __shfl(s, m);
    int   jm = __shfl(j, m);
    rank += (sm > s || (sm == s && jm < j)) ? 1 : 0;
  }
  bool sel = valid && (rank < W_TOP);
  unsigned mask = (unsigned)__ballot(sel);
  int pos = __popc(mask & ((1u << lane) - 1u));   // ascending-index compaction
  if (sel) ids[row * W_TOP + pos] = j;
}

// ---------------------------------------------------------------------------
// Kernel 3: fused 12-step GRU. One WG = 16 batch rows, all 512 hidden dims,
// all 12 timesteps (recurrence is row-local => no inter-WG sync).
// 8 waves x 4 column-groups; each wave keeps 6 fp32 WMMA accumulators
// (zero-initialized; biases folded into the gate epilogue) and the fp32
// carry state h in registers. LDS holds f16 GEMM operands: gathered x_t and
// ping-pong h (t&1 indexing, compile-time constant after unroll-by-2).
// ---------------------------------------------------------------------------
__global__ __launch_bounds__(256, 1) void gru_kernel(
    const float* __restrict__ x, const int* __restrict__ ids,
    const _Float16* __restrict__ wiP, const _Float16* __restrict__ whP,
    const float* __restrict__ bih, const float* __restrict__ bhh,
    float* __restrict__ out) {
  __shared__ _Float16 xbuf[16 * LDST];
  __shared__ _Float16 hbuf[2][16 * LDST];

  int tid  = threadIdx.x;
  int wave = tid >> 5;
  int lane = tid & 31;
  int g    = lane >> 4;
  int ln   = lane & 15;
  int wgBase = blockIdx.x * 16;           // first batch row of this WG

  // zero initial hidden state (f16 operand buffer, read at t=0)
  for (int i = tid; i < 16 * LDST; i += 256) hbuf[0][i] = (_Float16)0.f;

  // hoisted per-lane bases
  const _Float16* axb = xbuf + ln * LDST + g * 8;          // A-frag base (x)
  const int laneOff = lane * 16;                           // packed-B lane offset

  // per-lane epilogue biases (r/z gates always see b_ih+b_hh summed) + carry
  float brz[4], bzz[4], bin[4], bhn[4];
  v8f hreg[4];
#pragma unroll
  for (int jj = 0; jj < 4; ++jj) {
    int col = (wave * 4 + jj) * 16 + ln;
    brz[jj] = bih[col] + bhh[col];
    bzz[jj] = bih[D_DIM + col] + bhh[D_DIM + col];
    bin[jj] = bih[2 * D_DIM + col];
    bhn[jj] = bhh[2 * D_DIM + col];
    hreg[jj] = v8f{};
  }
  __syncthreads();

#pragma unroll 2
  for (int t = 0; t < 12; ++t) {
    // --- gather x_t rows (fp32 global -> f16 LDS), 64B of halves per thread
    {
      int m   = tid >> 4;               // 0..15 row within WG
      int c   = (tid & 15) * 32;        // 32-elem chunk
      int row = wgBase + m;
      int id  = ids[row * W_TOP + t];
      const float* src = x + ((size_t)(row >> 9) * T_SEQ + id) * D_DIM + c;
      _Float16* dst = xbuf + m * LDST + c;
#pragma unroll
      for (int u = 0; u < 32; u += 4) {
        float4 v = *(const float4*)(src + u);
        dst[u + 0] = (_Float16)v.x; dst[u + 1] = (_Float16)v.y;
        dst[u + 2] = (_Float16)v.z; dst[u + 3] = (_Float16)v.w;
      }
    }
    __syncthreads();

    // ping-pong: t&1 is a compile-time constant in each unrolled copy
    const _Float16* ahb = hbuf[t & 1] + ln * LDST + g * 8; // A-frag base (h)
    _Float16*       hn  = hbuf[(t & 1) ^ 1];

    for (int jj = 0; jj < 4; ++jj) {
      int j   = wave * 4 + jj;          // 0..31 -> ct = j / 32+j / 64+j
      int col = j * 16 + ln;
      // hoisted per-lane fragment bases; kk advances by immediate offsets only
      const _Float16* pb0 = wiP + ((size_t)(j)      << 13) + laneOff;
      const _Float16* pb1 = wiP + ((size_t)(32 + j) << 13) + laneOff;
      const _Float16* pb2 = wiP + ((size_t)(64 + j) << 13) + laneOff;
      const _Float16* pc0 = whP + ((size_t)(j)      << 13) + laneOff;
      const _Float16* pc1 = whP + ((size_t)(32 + j) << 13) + laneOff;
      const _Float16* pc2 = whP + ((size_t)(64 + j) << 13) + laneOff;

      v8f air = v8f{}, aiz = v8f{}, ain = v8f{};
      v8f ahr = v8f{}, ahz = v8f{}, ahn = v8f{};

#pragma unroll
      for (int kk = 0; kk < NKC; ++kk) {
        int k0 = kk << 5;               // A offset (halves)
        int w0 = kk << 9;               // packed-B offset (halves) = 1KB/chunk
        v16h ax = ldA(axb, k0);
        v16h ah = ldA(ahb, k0);
        v16h b0 = *(const v16h*)(pb0 + w0);
        v16h b1 = *(const v16h*)(pb1 + w0);
        v16h b2 = *(const v16h*)(pb2 + w0);
        air = __builtin_amdgcn_wmma_f32_16x16x32_f16(false, ax, false, b0, (short)0, air, false, false);
        aiz = __builtin_amdgcn_wmma_f32_16x16x32_f16(false, ax, false, b1, (short)0, aiz, false, false);
        ain = __builtin_amdgcn_wmma_f32_16x16x32_f16(false, ax, false, b2, (short)0, ain, false, false);
        v16h c0 = *(const v16h*)(pc0 + w0);
        v16h c1 = *(const v16h*)(pc1 + w0);
        v16h c2 = *(const v16h*)(pc2 + w0);
        ahr = __builtin_amdgcn_wmma_f32_16x16x32_f16(false, ah, false, c0, (short)0, ahr, false, false);
        ahz = __builtin_amdgcn_wmma_f32_16x16x32_f16(false, ah, false, c1, (short)0, ahz, false, false);
        ahn = __builtin_amdgcn_wmma_f32_16x16x32_f16(false, ah, false, c2, (short)0, ahn, false, false);
        if (kk + 2 < NKC) {
          // speculative prefetch 2 chunks ahead on the r-gate streams
          __builtin_prefetch(pb0 + w0 + 1024, 0, 1);
          __builtin_prefetch(pc0 + w0 + 1024, 0, 1);
        }
      }

      // fused GRU gates, fp32, in registers (C/D tile: element i -> M = g*8+i)
#pragma unroll
      for (int i = 0; i < 8; ++i) {
        float r = 1.f / (1.f + __expf(-(air[i] + ahr[i] + brz[jj])));
        float z = 1.f / (1.f + __expf(-(aiz[i] + ahz[i] + bzz[jj])));
        float n = tanhf(ain[i] + bin[jj] + r * (ahn[i] + bhn[jj]));
        float h = (1.f - z) * n + z * hreg[jj][i];
        hreg[jj][i] = h;                       // fp32 carry stays in VGPRs
        int m = g * 8 + i;
        hn[m * LDST + col] = (_Float16)h;      // f16 operand for next step
      }
    }
    __syncthreads();
  }

  // final hidden state -> output (fp32), row-major (b, tq_idx, d)
#pragma unroll
  for (int jj = 0; jj < 4; ++jj) {
    int col = (wave * 4 + jj) * 16 + ln;
#pragma unroll
    for (int i = 0; i < 8; ++i) {
      int m = g * 8 + i;
      out[(size_t)(wgBase + m) * D_DIM + col] = hreg[jj][i];
    }
  }
}

// ---------------------------------------------------------------------------
extern "C" void kernel_launch(void* const* d_in, const int* in_sizes, int n_in,
                              void* d_out, int out_size, void* d_ws, size_t ws_size,
                              hipStream_t stream) {
  const float* x   = (const float*)d_in[0];
  const float* wih = (const float*)d_in[1];
  const float* whh = (const float*)d_in[2];
  const float* bih = (const float*)d_in[3];
  const float* bhh = (const float*)d_in[4];
  float* out = (float*)d_out;

  // workspace layout (256B aligned): packed w_ih16 | packed w_hh16 | ids
  char* ws = (char*)d_ws;
  _Float16* wiP = (_Float16*)(ws);
  _Float16* whP = (_Float16*)(ws + 1572864);
  int*      ids = (int*)(ws + 3145728);       // 8192 * 12 ints

  pack_w_kernel<<<(NCT * NKC * 32 + 255) / 256, 256, 0, stream>>>(wih, whh, wiP, whP);
  topk_kernel<<<NROW / 8, 256, 0, stream>>>(x, ids);
  gru_kernel<<<NROW / 16, 256, 0, stream>>>(x, ids, wiP, whP, bih, bhh, out);
}
